// HAD_GNN_25237227831863
// MI455X (gfx1250) — matrix operand order repaired and checked by reading, not compiled, not run on hardware
//
#include <hip/hip_runtime.h>

#define T_LAG 10
#define IN_DIM 12
#define HID 64
#define GOUT 32
#define NCLS 3
#define KPAD 96           // 12 + 64 padded to 3 x 32
#define Z4H 256           // 4*HID gate width
#define NFRAG 48          // 3 k-steps x 16 column tiles

typedef __attribute__((ext_vector_type(16))) _Float16 v16h;
typedef __attribute__((ext_vector_type(8)))  float    v8f;

union AFrag { v16h v; _Float16 e[16]; };
union CFrag { v8f  v; float    e[8];  };

// ---------------------------------------------------------------------------
// Prep: bsum = bih + bhh, and pack the concatenated LSTM weight [Wih^T;Whh^T;0]
// (96 x 256) directly into per-lane WMMA B-fragment layout:
//   Bpack[frag = kt*16+ct][lane 0..31][half 0..15], 32 bytes per lane, so each
//   lane fetches a whole fragment with two contiguous b128 loads.
// B 32x16 lane layout: n = lane&15; k = (h&7) + (lane>=16 ? 8 : 0) + (h>=8 ? 16 : 0)
// ---------------------------------------------------------------------------
__global__ void prep_pack(const float* __restrict__ Wih, const float* __restrict__ Whh,
                          const float* __restrict__ bih, const float* __restrict__ bhh,
                          _Float16* __restrict__ Bpack, float* __restrict__ bsum) {
    int tid = blockIdx.x * blockDim.x + threadIdx.x;
    for (int idx = tid; idx < NFRAG * 32 * 16; idx += blockDim.x * gridDim.x) {
        int h    = idx & 15;
        int lane = (idx >> 4) & 31;
        int f    = idx >> 9;
        int kt = f / 16, ct = f % 16;
        int nn = lane & 15;
        int kk = (h & 7) + ((lane >> 4) ? 8 : 0) + ((h >> 3) ? 16 : 0);
        int k = kt * 32 + kk;           // row of Wcat (input-dim axis)
        int j = ct * 16 + nn;           // column (gate axis, 0..255)
        float v = 0.f;
        if (k < IN_DIM)            v = Wih[j * IN_DIM + k];
        else if (k < IN_DIM + HID) v = Whh[j * HID + (k - IN_DIM)];
        Bpack[idx] = (_Float16)v;
    }
    for (int j = tid; j < Z4H; j += blockDim.x * gridDim.x) bsum[j] = bih[j] + bhh[j];
}

// ---------------------------------------------------------------------------
// Fused LSTM + temporal attention + FC. One block (8 waves) per 16-node tile.
// B fragments (weights) are loaded ONCE per wave and stay in VGPRs across the
// whole time loop; per timestep only the A tile is staged (pre-swizzled into
// per-lane fragment layout in LDS so reads are ds_load_b128).
// ---------------------------------------------------------------------------
__global__ __launch_bounds__(256)
void lstm_att_fc(const float* __restrict__ x, const _Float16* __restrict__ Bpack,
                 const float* __restrict__ bsum, const float* __restrict__ tatt,
                 const float* __restrict__ fcW, const float* __restrict__ fcb,
                 float* __restrict__ feat, int n) {
    __shared__ __align__(32) _Float16 xhsw[3][32][16]; //  3 KB  A fragments, pre-swizzled
    __shared__ float    zz[16][Z4H];                   // 16 KB  gate pre-activations
    __shared__ float    hh[16][HID];                   //  4 KB  h state (f32)
    __shared__ float    cs[16][HID];                   //  4 KB  c state, reused as att_ht
    __shared__ _Float16 hts[T_LAG][16][HID];           // 20 KB  h history for attention
    __shared__ float    sc[16][T_LAG];

    const int tid    = threadIdx.x;
    const int lane   = tid & 31;
    const int wave   = tid >> 5;
    const int m16    = lane & 15;
    const int laneHi = lane >> 4;
    const int nodeBase = blockIdx.x * 16;

    // Resident B fragments: 2 column tiles (wave, wave+8) x 3 k-steps
    AFrag bfr[2][3];
    #pragma unroll
    for (int cc = 0; cc < 2; ++cc) {
        int ct = wave + cc * 8;
        #pragma unroll
        for (int kt = 0; kt < 3; ++kt) {
            const v16h* src = (const v16h*)(Bpack + (((kt * 16 + ct) * 32) + lane) * 16);
            bfr[cc][kt].v = *src;
        }
    }
    float bias0 = bsum[wave * 16 + m16];
    float bias1 = bsum[(wave + 8) * 16 + m16];

    for (int idx = tid; idx < 16 * HID; idx += 256) {
        hh[idx >> 6][idx & 63] = 0.f; cs[idx >> 6][idx & 63] = 0.f;
    }
    __syncthreads();

    for (int t = 0; t < T_LAG; ++t) {
        // Stage A = [x_t | h | pad] as f16, directly in WMMA A-fragment layout:
        // element (m, k): kt=k>>5, kk=k&31; laneHi=(kk>>3)&1; upper=(kk>>4)&1;
        // base=kk&7; vg=upper*4+(base>>1); h=2*vg+(base&1); lane=m+16*laneHi.
        for (int idx = tid; idx < 16 * KPAD; idx += 256) {
            int m = idx / KPAD, k = idx % KPAD;
            int node = nodeBase + m;
            float v = 0.f;
            if (k < IN_DIM) { if (node < n) v = x[((long long)node * T_LAG + t) * IN_DIM + k]; }
            else if (k < IN_DIM + HID) v = hh[m][k - IN_DIM];
            int kt = k >> 5, kk = k & 31;
            int lHi = (kk >> 3) & 1, up = (kk >> 4) & 1, base = kk & 7;
            int h = 2 * (up * 4 + (base >> 1)) + (base & 1);
            xhsw[kt][m + 16 * lHi][h] = (_Float16)v;
        }
        __syncthreads();

        // GEMM: two 16-col tiles per wave, A shared between them per k-step
        CFrag acc0, acc1;
        #pragma unroll
        for (int i = 0; i < 8; ++i) { acc0.e[i] = 0.f; acc1.e[i] = 0.f; }
        #pragma unroll
        for (int kt = 0; kt < 3; ++kt) {
            AFrag a;
            a.v = *(const v16h*)&xhsw[kt][lane][0];
            acc0.v = __builtin_amdgcn_wmma_f32_16x16x32_f16(
                false, a.v, false, bfr[0][kt].v, (short)0, acc0.v, false, false);
            acc1.v = __builtin_amdgcn_wmma_f32_16x16x32_f16(
                false, a.v, false, bfr[1][kt].v, (short)0, acc1.v, false, false);
        }
        #pragma unroll
        for (int vg = 0; vg < 8; ++vg) {
            int m = vg + (laneHi ? 8 : 0);
            zz[m][wave * 16 + m16]       = acc0.e[vg] + bias0;
            zz[m][(wave + 8) * 16 + m16] = acc1.e[vg] + bias1;
        }
        __syncthreads();

        // Gates (torch order i,f,g,o), f32 state update
        for (int idx = tid; idx < 16 * HID; idx += 256) {
            int m = idx >> 6, j = idx & 63;
            float zi = zz[m][j], zf = zz[m][64 + j], zg = zz[m][128 + j], zo = zz[m][192 + j];
            float ig = 1.f / (1.f + __expf(-zi));
            float fg = 1.f / (1.f + __expf(-zf));
            float gg = tanhf(zg);
            float og = 1.f / (1.f + __expf(-zo));
            float cn = fg * cs[m][j] + ig * gg;
            float hn = og * tanhf(cn);
            cs[m][j] = cn; hh[m][j] = hn;
            hts[t][m][j] = (_Float16)hn;
        }
        __syncthreads();

        if (tid < 16) {
            float s = 0.f;
            for (int j = 0; j < HID; ++j) s += hh[tid][j] * tatt[t * HID + j];
            sc[tid][t] = s;
        }
        __syncthreads();
    }

    // Temporal softmax per node
    if (tid < 16) {
        float mx = sc[tid][0];
        for (int t = 1; t < T_LAG; ++t) mx = fmaxf(mx, sc[tid][t]);
        float den = 0.f;
        for (int t = 0; t < T_LAG; ++t) { float e = __expf(sc[tid][t] - mx); sc[tid][t] = e; den += e; }
        float inv = 1.f / den;
        for (int t = 0; t < T_LAG; ++t) sc[tid][t] *= inv;
    }
    __syncthreads();

    // att_ht (into cs)
    for (int idx = tid; idx < 16 * HID; idx += 256) {
        int m = idx >> 6, j = idx & 63;
        float s = 0.f;
        for (int t = 0; t < T_LAG; ++t) s += sc[m][t] * (float)hts[t][m][j];
        cs[m][j] = s;
    }
    __syncthreads();

    // fc + ReLU -> feat
    for (int idx = tid; idx < 16 * HID; idx += 256) {
        int m = idx >> 6, jo = idx & 63;
        int node = nodeBase + m;
        if (node >= n) continue;
        float acc = fcb[jo];
        for (int k = 0; k < HID; ++k) acc += cs[m][k] * fcW[jo * HID + k];
        feat[(long long)node * HID + jo] = fmaxf(acc, 0.f);
    }
}

// ---------------------------------------------------------------------------
// GAT projection: h = [f1 | f2] @ W, plus per-node attention dots.
// One wave per node (8 nodes / 256-thread block).
// ---------------------------------------------------------------------------
__global__ __launch_bounds__(256)
void gat_project(const float* __restrict__ f1, int d1,
                 const float* __restrict__ f2, int s2, int o2, int d2,
                 const float* __restrict__ W, const float* __restrict__ asrc,
                 const float* __restrict__ adst,
                 float* __restrict__ hbuf, float* __restrict__ as_, float* __restrict__ ad_,
                 int n) {
    int node = blockIdx.x * 8 + (threadIdx.x >> 5);
    int j = threadIdx.x & 31;
    if (node >= n) return;
    float acc = 0.f;
    for (int k = 0; k < d1; ++k) acc += f1[(long long)node * d1 + k] * W[k * GOUT + j];
    for (int k = 0; k < d2; ++k) acc += f2[(long long)node * s2 + o2 + k] * W[(d1 + k) * GOUT + j];
    hbuf[(long long)node * GOUT + j] = acc;
    float va = acc * asrc[j], vd = acc * adst[j];
    for (int off = 16; off; off >>= 1) {
        va += __shfl_down(va, off, 32);
        vd += __shfl_down(vd, off, 32);
    }
    if (j == 0) { as_[node] = va; ad_[node] = vd; }
}

__global__ void seg_init(float* __restrict__ emax, float* __restrict__ denom,
                         float* __restrict__ num, int n) {
    int idx = blockIdx.x * blockDim.x + threadIdx.x;
    if (idx < n) { emax[idx] = -3.0e38f; denom[idx] = 0.f; }
    if (idx < n * GOUT) num[idx] = 0.f;
}

// Pass 1: per-destination segment max of leaky_relu(as[src]+ad[dst]) (self loops appended)
__global__ void edge_max(const int* __restrict__ src, const int* __restrict__ dst,
                         const float* __restrict__ as_, const float* __restrict__ ad_,
                         float* __restrict__ emax, int E, int n) {
    int e = blockIdx.x * blockDim.x + threadIdx.x;
    if (e >= E + n) return;
    int s, d;
    if (e < E) { s = src[e]; d = dst[e]; } else { s = e - E; d = s; }
    float ee = as_[s] + ad_[d];
    ee = ee > 0.f ? ee : 0.2f * ee;
    if (ee >= 0.f) atomicMax((int*)(emax + d), __float_as_int(ee));
    else           atomicMin((unsigned int*)(emax + d), (unsigned int)__float_as_int(ee));
}

// Pass 2: 32 lanes per edge accumulate exp-weighted numerator + denominator
__global__ void edge_acc(const int* __restrict__ src, const int* __restrict__ dst,
                         const float* __restrict__ as_, const float* __restrict__ ad_,
                         const float* __restrict__ emax, const float* __restrict__ hbuf,
                         float* __restrict__ denom, float* __restrict__ num, int E, int n) {
    long long gid = (long long)blockIdx.x * blockDim.x + threadIdx.x;
    int e = (int)(gid >> 5);
    int k = (int)(gid & 31);
    if (e >= E + n) return;
    int s, d;
    if (e < E) { s = src[e]; d = dst[e]; } else { s = e - E; d = s; }
    float ee = as_[s] + ad_[d];
    ee = ee > 0.f ? ee : 0.2f * ee;
    float ex = __expf(ee - emax[d]);
    if (k == 0) atomicAdd(denom + d, ex);
    atomicAdd(num + (long long)d * GOUT + k, ex * hbuf[(long long)s * GOUT + k]);
}

__global__ void gat_finalize(float* __restrict__ num, const float* __restrict__ denom,
                             const float* __restrict__ b, int n) {
    int idx = blockIdx.x * blockDim.x + threadIdx.x;
    if (idx >= n * GOUT) return;
    int nn = idx >> 5, k = idx & 31;
    num[idx] = num[idx] / (denom[nn] + 1e-16f) + b[k];
}

__global__ void final_logits(const float* __restrict__ h2, const float* __restrict__ linW,
                             const float* __restrict__ linb, float* __restrict__ out, int n) {
    int node = blockIdx.x * blockDim.x + threadIdx.x;
    if (node >= n) return;
    float l[NCLS];
    #pragma unroll
    for (int c = 0; c < NCLS; ++c) {
        float acc = linb[c];
        for (int k = 0; k < GOUT; ++k) acc += h2[(long long)node * GOUT + k] * linW[c * GOUT + k];
        l[c] = fmaxf(acc, 0.f);
    }
    float mx = fmaxf(l[0], fmaxf(l[1], l[2]));
    float den = 0.f;
    #pragma unroll
    for (int c = 0; c < NCLS; ++c) den += __expf(l[c] - mx);
    float lse = mx + __logf(den);
    #pragma unroll
    for (int c = 0; c < NCLS; ++c) out[node * NCLS + c] = l[c] - lse;
}

// ---------------------------------------------------------------------------
extern "C" void kernel_launch(void* const* d_in, const int* in_sizes, int n_in,
                              void* d_out, int out_size, void* d_ws, size_t ws_size,
                              hipStream_t stream) {
    const float* x        = (const float*)d_in[0];
    const int*   edges    = (const int*)d_in[1];
    const float* lstm_Wih = (const float*)d_in[2];
    const float* lstm_Whh = (const float*)d_in[3];
    const float* lstm_bih = (const float*)d_in[4];
    const float* lstm_bhh = (const float*)d_in[5];
    const float* tatt_W   = (const float*)d_in[6];
    const float* fc_W     = (const float*)d_in[7];
    const float* fc_b     = (const float*)d_in[8];
    const float* gat1_W   = (const float*)d_in[9];
    const float* gat1_as  = (const float*)d_in[10];
    const float* gat1_ad  = (const float*)d_in[11];
    const float* gat1_b   = (const float*)d_in[12];
    const float* gat2_W   = (const float*)d_in[13];
    const float* gat2_as  = (const float*)d_in[14];
    const float* gat2_ad  = (const float*)d_in[15];
    const float* gat2_b   = (const float*)d_in[16];
    const float* lin_W    = (const float*)d_in[17];
    const float* lin_b    = (const float*)d_in[18];
    float* out = (float*)d_out;

    const int n = in_sizes[0] / (T_LAG * IN_DIM);
    const int E = in_sizes[1] / 2;
    const int* esrc = edges;
    const int* edst = edges + E;

    // Workspace carve-up (256B aligned)
    char* p = (char*)d_ws;
    auto carve = [&](size_t bytes) { char* r = p; p += (bytes + 255) & ~size_t(255); return r; };
    _Float16* Bpack = (_Float16*)carve((size_t)NFRAG * 32 * 16 * sizeof(_Float16));
    float* bsum  = (float*)carve(Z4H * sizeof(float));
    float* feat  = (float*)carve((size_t)n * HID * sizeof(float));
    float* hbuf  = (float*)carve((size_t)n * GOUT * sizeof(float));
    float* as_   = (float*)carve((size_t)n * sizeof(float));
    float* ad_   = (float*)carve((size_t)n * sizeof(float));
    float* emax  = (float*)carve((size_t)n * sizeof(float));
    float* denom = (float*)carve((size_t)n * sizeof(float));
    float* num1  = (float*)carve((size_t)n * GOUT * sizeof(float));
    float* num2  = (float*)carve((size_t)n * GOUT * sizeof(float));

    // 1) LSTM weights -> f16, packed in WMMA fragment layout
    prep_pack<<<96, 256, 0, stream>>>(lstm_Wih, lstm_Whh, lstm_bih, lstm_bhh, Bpack, bsum);

    // 2) Fused LSTM + attention + FC (WMMA, resident weight fragments)
    lstm_att_fc<<<(n + 15) / 16, 256, 0, stream>>>(x, Bpack, bsum, tatt_W, fc_W, fc_b, feat, n);

    const int segBlocks  = (n * GOUT + 255) / 256;
    const int edgeBlocks = (E + n + 255) / 256;
    const long long accThreads = (long long)(E + n) * 32;
    const int accBlocks  = (int)((accThreads + 255) / 256);

    // 3) GAT layer 1
    gat_project<<<(n + 7) / 8, 256, 0, stream>>>(feat, HID, feat, 0, 0, 0,
                                                 gat1_W, gat1_as, gat1_ad,
                                                 hbuf, as_, ad_, n);
    seg_init<<<segBlocks, 256, 0, stream>>>(emax, denom, num1, n);
    edge_max<<<edgeBlocks, 256, 0, stream>>>(esrc, edst, as_, ad_, emax, E, n);
    edge_acc<<<accBlocks, 256, 0, stream>>>(esrc, edst, as_, ad_, emax, hbuf, denom, num1, E, n);
    gat_finalize<<<segBlocks, 256, 0, stream>>>(num1, denom, gat1_b, n);

    // 4) GAT layer 2: input = [h1 | x[:, T-1, :]]
    gat_project<<<(n + 7) / 8, 256, 0, stream>>>(num1, GOUT, x, T_LAG * IN_DIM,
                                                 (T_LAG - 1) * IN_DIM, IN_DIM,
                                                 gat2_W, gat2_as, gat2_ad,
                                                 hbuf, as_, ad_, n);
    seg_init<<<segBlocks, 256, 0, stream>>>(emax, denom, num2, n);
    edge_max<<<edgeBlocks, 256, 0, stream>>>(esrc, edst, as_, ad_, emax, E, n);
    edge_acc<<<accBlocks, 256, 0, stream>>>(esrc, edst, as_, ad_, emax, hbuf, denom, num2, E, n);
    gat_finalize<<<segBlocks, 256, 0, stream>>>(num2, denom, gat2_b, n);

    // 5) Classifier + log_softmax
    final_logits<<<(n + 255) / 256, 256, 0, stream>>>(num2, lin_W, lin_b, out, n);
}
